// USSPA_G_T2_65326452572275
// MI455X (gfx1250) — matrix operand order repaired and checked by reading, not compiled, not run on hardware
//
#include <hip/hip_runtime.h>
#include <hip/hip_bf16.h>

typedef _Float16 half_t;
typedef __attribute__((ext_vector_type(16))) _Float16 v16h;
typedef __attribute__((ext_vector_type(8)))  _Float16 v8h;
typedef __attribute__((ext_vector_type(8)))  float    v8f;

#define B2   16      // 2*B batches
#define NPTS 2048    // encoder points
#define NP   512     // GPN

#define SHUF16(x, y)                                                     \
  __builtin_shufflevector((x), (y), 0, 1, 2, 3, 4, 5, 6, 7, 8, 9, 10,    \
                          11, 12, 13, 14, 15)

// ---------------------------------------------------------------------------
// Weight/bias padding: W f32 [M][K] -> f16 [Mp][Kp] (zeros beyond), bias -> [Mp]
// Row-major padded W is directly fragment-ready for the B operand.
// ---------------------------------------------------------------------------
__global__ __launch_bounds__(256)
void k_padw(const float* __restrict__ W, half_t* __restrict__ Wp,
            int M, int K, int Mp, int Kp) {
  long i = (long)blockIdx.x * 256 + threadIdx.x;
  if (i >= (long)Mp * Kp) return;
  int r = (int)(i / Kp), c = (int)(i % Kp);
  float w = 0.f;
  if (r < M && c < K) w = W[(long)r * K + c];
  Wp[i] = (half_t)w;
}

__global__ __launch_bounds__(256)
void k_padb(const float* __restrict__ B, float* __restrict__ Bp, int M, int Mp) {
  int i = blockIdx.x * 256 + threadIdx.x;
  if (i >= Mp) return;
  Bp[i] = (i < M) ? B[i] : 0.f;
}

// ---------------------------------------------------------------------------
// Pack [B,N,3] f32 inputs into point-major f16 x0 [16][2048][64] (ch 3.. zero)
// ---------------------------------------------------------------------------
__global__ __launch_bounds__(256)
void k_pack(const float* __restrict__ P, const float* __restrict__ C,
            half_t* __restrict__ X0) {
  int i = blockIdx.x * blockDim.x + threadIdx.x;
  if (i >= B2 * NPTS) return;
  int b = i / NPTS, n = i % NPTS;
  const float* s = (b < 8) ? (P + ((long)b * NPTS + n) * 3)
                           : (C + ((long)(b - 8) * NPTS + n) * 3);
  half_t* d = X0 + (long)i * 64;
  #pragma unroll
  for (int c = 0; c < 64; ++c)
    d[c] = (c < 3) ? (half_t)s[c] : (half_t)0.f;
}

// ---------------------------------------------------------------------------
// LDS-free WMMA GEMM on point-major activations, 2-deep software pipeline.
//   X[b][n][k] (pitch Kp, Kp % 64 == 0), W f16 [Mp][Kp] row-major padded,
//   bias f32 [Mp], Y[b][n][o] (pitch Mp);  D = X * W^T + bias
// Block: 64 points x 64 out-channels; 8 waves (4 point-stripes x 2 o-pairs);
// wave = two 16x16 C tiles sharing one A fragment. Two alternating fragment
// register sets: loads for one K-step issue before the other's WMMAs, so no
// rotation copies and global latency overlaps compute.
// Per K-step per wave: 6x global_load_b128 (L2-resident) + 2x wmma.
// ---------------------------------------------------------------------------
__global__ __launch_bounds__(256)
void k_gemm(const half_t* __restrict__ X, long xStride,
            const half_t* __restrict__ W, const float* __restrict__ bias,
            half_t* __restrict__ Y, long yStride,
            int Kp, int Mp, int relu) {
  const int b = blockIdx.z;
  const half_t* Xb = X + (long)b * xStride;
  half_t*       Yb = Y + (long)b * yStride;
  const int n0 = blockIdx.y * 64;            // point-tile base
  const int o0 = blockIdx.x * 64;            // out-channel-tile base
  const int lane = threadIdx.x & 31;
  const int wave = threadIdx.x >> 5;
  const int wm = wave >> 1;                  // 0..3 point stripe
  const int wn = wave & 1;                   // 0..1 o-tile pair

  const int prow = n0 + wm * 16 + (lane & 15);   // A row: my point
  const int ka = (lane >> 4) ? 16 : 0;           // A K-offset per lane half
  const int kb = (lane >> 4) ? 16 : 0;           // B K-offset per lane half
  const int oc0 = o0 + wn * 32 + (lane & 15);    // B col: my out channel, tile 0
  const int oc1 = oc0 + 16;                      // tile 1

  const half_t* arow  = Xb + (long)prow * Kp;
  const half_t* brow0 = W + (long)oc0 * Kp;
  const half_t* brow1 = W + (long)oc1 * Kp;
  const float bv0 = bias[oc0];
  const float bv1 = bias[oc1];

  // fragment register set S0 preloaded at K-step 0
  v8h a0 = *reinterpret_cast<const v8h*>(arow + ka);
  v8h a1 = *reinterpret_cast<const v8h*>(arow + 8 + ka);
  v8h c0 = *reinterpret_cast<const v8h*>(brow0 + kb);
  v8h c1 = *reinterpret_cast<const v8h*>(brow0 + kb + 8);
  v8h c2 = *reinterpret_cast<const v8h*>(brow1 + kb);
  v8h c3 = *reinterpret_cast<const v8h*>(brow1 + kb + 8);

  v8f acc0 = {}, acc1 = {};
  for (int kk = 0; kk < Kp; kk += 64) {
    // load set S1 at kk+32 (always in range: Kp % 64 == 0)
    const half_t* an1  = arow  + kk + 32;
    const half_t* bn10 = brow0 + kk + 32;
    const half_t* bn11 = brow1 + kk + 32;
    v8h d0 = *reinterpret_cast<const v8h*>(an1 + ka);
    v8h d1 = *reinterpret_cast<const v8h*>(an1 + 8 + ka);
    v8h e0 = *reinterpret_cast<const v8h*>(bn10 + kb);
    v8h e1 = *reinterpret_cast<const v8h*>(bn10 + kb + 8);
    v8h e2 = *reinterpret_cast<const v8h*>(bn11 + kb);
    v8h e3 = *reinterpret_cast<const v8h*>(bn11 + kb + 8);
    if (kk + 96 < Kp) {
      __builtin_prefetch(an1 + 64, 0, 1);
      __builtin_prefetch(bn10 + 64, 0, 1);
    }

    // WMMAs on S0 (operands loaded one K-step ago)
    {
      v16h a   = SHUF16(a0, a1);
      v16h bf0 = SHUF16(c0, c1);
      v16h bf1 = SHUF16(c2, c3);
      acc0 = __builtin_amdgcn_wmma_f32_16x16x32_f16(false, a, false, bf0,
                                                    (short)0, acc0, false, false);
      acc1 = __builtin_amdgcn_wmma_f32_16x16x32_f16(false, a, false, bf1,
                                                    (short)0, acc1, false, false);
    }

    // reload set S0 at kk+64 (skip on last iteration)
    if (kk + 64 < Kp) {
      const half_t* an0  = arow  + kk + 64;
      const half_t* bn00 = brow0 + kk + 64;
      const half_t* bn01 = brow1 + kk + 64;
      a0 = *reinterpret_cast<const v8h*>(an0 + ka);
      a1 = *reinterpret_cast<const v8h*>(an0 + 8 + ka);
      c0 = *reinterpret_cast<const v8h*>(bn00 + kb);
      c1 = *reinterpret_cast<const v8h*>(bn00 + kb + 8);
      c2 = *reinterpret_cast<const v8h*>(bn01 + kb);
      c3 = *reinterpret_cast<const v8h*>(bn01 + kb + 8);
      if (kk + 128 < Kp) {
        __builtin_prefetch(an0 + 64, 0, 1);
        __builtin_prefetch(bn00 + 64, 0, 1);
      }
    }

    // WMMAs on S1
    {
      v16h a   = SHUF16(d0, d1);
      v16h bf0 = SHUF16(e0, e1);
      v16h bf1 = SHUF16(e2, e3);
      acc0 = __builtin_amdgcn_wmma_f32_16x16x32_f16(false, a, false, bf0,
                                                    (short)0, acc0, false, false);
      acc1 = __builtin_amdgcn_wmma_f32_16x16x32_f16(false, a, false, bf1,
                                                    (short)0, acc1, false, false);
    }
  }

  // D layout: VGPR r -> point = r + 8*(lane/16) (+stripe), col = lane%16 -> oc
  #pragma unroll
  for (int r = 0; r < 8; ++r) {
    int pt = n0 + wm * 16 + (lane >> 4) * 8 + r;
    float v0 = acc0[r] + bv0;
    float v1 = acc1[r] + bv1;
    if (relu) { v0 = fmaxf(v0, 0.f); v1 = fmaxf(v1, 0.f); }
    Yb[(long)pt * Mp + oc0] = (half_t)v0;
    Yb[(long)pt * Mp + oc1] = (half_t)v1;
  }
}

// ---------------------------------------------------------------------------
// Channel max over points: G[b][c] = max_n X[b][n][c]
// grid.x = B2 * (C/64); block = 4 point-groups x 64 channels
// ---------------------------------------------------------------------------
__global__ __launch_bounds__(256)
void k_chanmax(const half_t* __restrict__ X, half_t* __restrict__ G,
               int N, int C) {
  int chunks = C >> 6;
  int b = blockIdx.x / chunks, ch = blockIdx.x % chunks;
  int c = ch * 64 + (threadIdx.x & 63);
  int g = threadIdx.x >> 6;                 // 0..3
  const half_t* Xb = X + (long)b * N * C;
  float m = -3.4e38f;
  for (int n = g; n < N; n += 4) m = fmaxf(m, (float)Xb[(long)n * C + c]);
  __shared__ float s[4][64];
  s[g][threadIdx.x & 63] = m;
  __syncthreads();
  if (g == 0) {
    int t = threadIdx.x & 63;
    m = fmaxf(fmaxf(s[0][t], s[1][t]), fmaxf(s[2][t], s[3][t]));
    G[(long)b * C + c] = (half_t)m;
  }
}

// D[b][n][2C] = concat(X[b][n][C], broadcast G[b][C])
__global__ __launch_bounds__(256)
void k_concat_bcast(const half_t* __restrict__ X, const half_t* __restrict__ G,
                    half_t* __restrict__ D, int C, int N, long total) {
  long i = (long)blockIdx.x * 256 + threadIdx.x;
  if (i >= total) return;
  int C2 = 2 * C;
  long per = (long)N * C2;
  int b = (int)(i / per);
  long rem = i - (long)b * per;
  int n = (int)(rem / C2), c = (int)(rem % C2);
  D[i] = (c < C) ? X[((long)b * N + n) * C + c] : G[(long)b * C + (c - C)];
}

// D[b][n][Cd]: ch [0,C1) from Xa (pitch pA), [C1,C1+C2) from Xb (pitch pB), pad 0
__global__ __launch_bounds__(256)
void k_concat2(const half_t* __restrict__ Xa, int pA, int C1,
               const half_t* __restrict__ Xb, int pB, int C2,
               half_t* __restrict__ D, int Cd, int N, long total) {
  long i = (long)blockIdx.x * 256 + threadIdx.x;
  if (i >= total) return;
  long per = (long)N * Cd;
  int b = (int)(i / per);
  long rem = i - (long)b * per;
  int n = (int)(rem / Cd), c = (int)(rem % Cd);
  half_t v = (half_t)0.f;
  if (c < C1)            v = Xa[((long)b * N + n) * pA + c];
  else if (c < C1 + C2)  v = Xb[((long)b * N + n) * pB + (c - C1)];
  D[i] = v;
}

// feat G[16][512] -> F[64][512] (batch -> GEMM point rows, padded to 64)
__global__ __launch_bounds__(256)
void k_featpack(const half_t* __restrict__ G, half_t* __restrict__ F) {
  int i = blockIdx.x * 256 + threadIdx.x;
  if (i >= 64 * 512) return;
  int b = i / 512;
  F[i] = (b < B2) ? G[i] : (half_t)0.f;
}

// coarse [64][1536] f16 -> point f32 [16][512][3], pt f16 [16][512][64] (padded)
__global__ __launch_bounds__(256)
void k_point(const half_t* __restrict__ CO, float* __restrict__ PTF,
             half_t* __restrict__ PT16) {
  int i = blockIdx.x * 256 + threadIdx.x;
  if (i >= B2 * NP) return;
  int b = i / NP, n = i % NP;
  half_t* d = PT16 + (long)i * 64;
  #pragma unroll
  for (int k = 0; k < 3; ++k) {
    half_t v = CO[(long)b * 1536 + n * 3 + k];
    PTF[(long)i * 3 + k] = (float)v;
    d[k] = v;
  }
  #pragma unroll
  for (int c = 3; c < 64; ++c) d[c] = (half_t)0.f;
}

// pointnet2 ball query: first 32 in-radius indices in order, pad with first hit
__global__ __launch_bounds__(256)
void k_ballquery(const float* __restrict__ PTF, int* __restrict__ IDX) {
  int q = blockIdx.x * 256 + threadIdx.x;
  if (q >= B2 * NP) return;
  int b = q / NP, n = q % NP;
  const float* P = PTF + (long)b * NP * 3;
  float qx = P[n * 3], qy = P[n * 3 + 1], qz = P[n * 3 + 2];
  int* out = IDX + (long)q * 32;
  const float r2 = 0.25f * 0.25f;
  int cnt = 0;
  for (int j = 0; j < NP; ++j) {
    float dx = P[j * 3] - qx, dy = P[j * 3 + 1] - qy, dz = P[j * 3 + 2] - qz;
    if (dx * dx + dy * dy + dz * dz <= r2) {
      out[cnt++] = j;
      if (cnt >= 32) break;
    }
  }
  if (cnt == 0) { out[0] = n; cnt = 1; }        // query always qualifies
  for (int s = cnt; s < 32; ++s) out[s] = out[0];
}

// XL[b][n][c] = max over 32 neighbors of X2[b][idx][c]  (contiguous channels)
__global__ __launch_bounds__(256)
void k_gathermax(const half_t* __restrict__ X2, const int* __restrict__ IDX,
                 half_t* __restrict__ XL) {
  long i = (long)blockIdx.x * 256 + threadIdx.x;
  long total = (long)B2 * NP * 256;
  if (i >= total) return;
  int c = (int)(i & 255);
  long t = i >> 8;
  int n = (int)(t % NP), b = (int)(t / NP);
  const int* id = IDX + ((long)b * NP + n) * 32;
  const half_t* base = X2 + (long)b * NP * 256;
  float m = -3.4e38f;
  #pragma unroll 4
  for (int s = 0; s < 32; ++s) m = fmaxf(m, (float)base[(long)id[s] * 256 + c]);
  XL[i] = (half_t)m;
}

// out[b][n*4+u][k] = point[b][n][k] + shift[b][n][u*3+k]  (shift pitch 64)
__global__ __launch_bounds__(256)
void k_final(const float* __restrict__ PTF, const half_t* __restrict__ SH,
             float* __restrict__ out) {
  int i = blockIdx.x * 256 + threadIdx.x;
  if (i >= B2 * NP * 4 * 3) return;
  int k = i % 3, u = (i / 3) % 4, n = (i / 12) % NP, b = i / (12 * NP);
  float s = (float)SH[((long)b * NP + n) * 64 + (u * 3 + k)];
  out[i] = PTF[((long)b * NP + n) * 3 + k] + s;
}

// ---------------------------------------------------------------------------
extern "C" void kernel_launch(void* const* d_in, const int* in_sizes, int n_in,
                              void* d_out, int out_size, void* d_ws, size_t ws_size,
                              hipStream_t stream) {
  (void)in_sizes; (void)n_in; (void)out_size; (void)ws_size;
  const float* inP = (const float*)d_in[0];
  const float* inC = (const float*)d_in[1];

  char* ws = (char*)d_ws;
  half_t* ACT0 = (half_t*)ws;                       // 36 MB arena
  half_t* ACT1 = (half_t*)(ws + (36ull << 20));     // 34 MB arena
  char*   SM   = ws + (70ull << 20);                // small buffers (~23 MB)
  half_t* X0     = (half_t*)SM;                          // [16][2048][64] 4MB
  half_t* G16    = (half_t*)(SM + 0x400000ull);          // [16][<=512]
  half_t* FEAT   = (half_t*)(SM + 0x410000ull);          // [64][512]
  half_t* DA0    = (half_t*)(SM + 0x430000ull);          // [64][<=1536]
  half_t* DA1    = (half_t*)(SM + 0x470000ull);
  half_t* COARSE = (half_t*)(SM + 0x4B0000ull);          // [64][1536]
  float*  PTF    = (float*)(SM + 0x4F0000ull);           // [16][512][3] f32
  half_t* PT16   = (half_t*)(SM + 0x570000ull);          // [16][512][64] 1MB
  int*    IDX    = (int*)(SM + 0x670000ull);             // [16][512][32]
  half_t* SHIFT  = (half_t*)(SM + 0x8B0000ull);          // [16][512][64] 1MB
  half_t* WPOOL  = (half_t*)(SM + 0x9B0000ull);          // padded f16 weights
  float*  BPOOL  = (float*)(SM + 0x15B0000ull);          // padded f32 biases

  long woff = 0; int boff = 0;
  auto pad = [&](int widx, int M, int K, int Mp, int Kp, half_t** wout, float** bout) {
    half_t* wp = WPOOL + woff;
    float*  bp = BPOOL + boff;
    long tot = (long)Mp * Kp;
    k_padw<<<(int)((tot + 255) / 256), 256, 0, stream>>>(
        (const float*)d_in[widx], wp, M, K, Mp, Kp);
    k_padb<<<(Mp + 255) / 256, 256, 0, stream>>>(
        (const float*)d_in[widx + 1], bp, M, Mp);
    woff += tot; boff += Mp;
    *wout = wp; *bout = bp;
  };

  half_t *We1a, *We1b, *We2a, *We2b, *Wd0, *Wd1, *Wd2, *Wd3, *Wd4;
  half_t *Wr1a, *Wr1b, *Wr1c, *Wr2a, *Wr2b, *Wr2c, *Wr3a, *Wr3b, *Wr3c;
  half_t *Wr4a, *Wr4b, *Wr4c;
  float *Be1a, *Be1b, *Be2a, *Be2b, *Bd0, *Bd1, *Bd2, *Bd3, *Bd4;
  float *Br1a, *Br1b, *Br1c, *Br2a, *Br2b, *Br2c, *Br3a, *Br3b, *Br3c;
  float *Br4a, *Br4b, *Br4c;
  pad(2, 128, 3, 128, 64, &We1a, &Be1a);
  pad(4, 256, 128, 256, 128, &We1b, &Be1b);
  pad(6, 512, 512, 512, 512, &We2a, &Be2a);
  pad(8, 512, 512, 512, 512, &We2b, &Be2b);
  pad(10, 512, 512, 512, 512, &Wd0, &Bd0);
  pad(12, 512, 512, 512, 512, &Wd1, &Bd1);
  pad(14, 1024, 512, 1024, 512, &Wd2, &Bd2);
  pad(16, 1024, 1024, 1024, 1024, &Wd3, &Bd3);
  pad(18, 1536, 1024, 1536, 1024, &Wd4, &Bd4);
  pad(20, 256, 3, 256, 64, &Wr1a, &Br1a);
  pad(22, 256, 256, 256, 256, &Wr1b, &Br1b);
  pad(24, 256, 256, 256, 256, &Wr1c, &Br1c);
  pad(26, 256, 512, 256, 512, &Wr2a, &Br2a);
  pad(28, 256, 256, 256, 256, &Wr2b, &Br2b);
  pad(30, 256, 256, 256, 256, &Wr2c, &Br2c);
  pad(32, 256, 512, 256, 512, &Wr3a, &Br3a);
  pad(34, 256, 256, 256, 256, &Wr3b, &Br3b);
  pad(36, 256, 256, 256, 256, &Wr3c, &Br3c);
  pad(38, 512, 259, 512, 320, &Wr4a, &Br4a);
  pad(40, 512, 512, 512, 512, &Wr4b, &Br4b);
  pad(42, 12, 512, 64, 512, &Wr4c, &Br4c);

  // gemm on point-major activations: grid (Mp/64, pts/64, batches)
  auto gemm = [&](const half_t* X, long xs, const half_t* Wp, const float* Bp,
                  half_t* Y, long ys, int Mp, int Kp, int pts, int batches,
                  int relu) {
    dim3 g(Mp / 64, pts / 64, batches);
    k_gemm<<<g, 256, 0, stream>>>(X, xs, Wp, Bp, Y, ys, Kp, Mp, relu);
  };

  k_pack<<<(B2 * NPTS + 255) / 256, 256, 0, stream>>>(inP, inC, X0);

  // ---- encoder 1: 3(->64) -> 128 (relu) -> 256 ----
  half_t* E1A = ACT0;                                  // [16][2048][128]
  half_t* F1  = ACT1;                                  // [16][2048][256]
  gemm(X0, 64L * NPTS, We1a, Be1a, E1A, 128L * NPTS, 128, 64, NPTS, B2, 1);
  gemm(E1A, 128L * NPTS, We1b, Be1b, F1, 256L * NPTS, 256, 128, NPTS, B2, 0);
  k_chanmax<<<B2 * (256 / 64), 256, 0, stream>>>(F1, G16, NPTS, 256);
  half_t* CAT1 = ACT0;                                 // [16][2048][512]
  {
    long total = (long)B2 * NPTS * 512;
    k_concat_bcast<<<(int)((total + 255) / 256), 256, 0, stream>>>(
        F1, G16, CAT1, 256, NPTS, total);
  }
  // ---- encoder 2: 512 -> 512 (relu) -> 512 ----
  half_t* E2A = ACT1;
  gemm(CAT1, 512L * NPTS, We2a, Be2a, E2A, 512L * NPTS, 512, 512, NPTS, B2, 1);
  half_t* E2B = ACT0;
  gemm(E2A, 512L * NPTS, We2b, Be2b, E2B, 512L * NPTS, 512, 512, NPTS, B2, 0);
  k_chanmax<<<B2 * (512 / 64), 256, 0, stream>>>(E2B, G16, NPTS, 512);
  k_featpack<<<(64 * 512 + 255) / 256, 256, 0, stream>>>(G16, FEAT);

  // ---- D_A coarse decoder (batch as point rows, padded to 64) ----
  gemm(FEAT, 0, Wd0, Bd0, DA0, 0, 512, 512, 64, 1, 1);
  gemm(DA0, 0, Wd1, Bd1, DA1, 0, 512, 512, 64, 1, 1);
  gemm(DA1, 0, Wd2, Bd2, DA0, 0, 1024, 512, 64, 1, 1);
  gemm(DA0, 0, Wd3, Bd3, DA1, 0, 1024, 1024, 64, 1, 1);
  gemm(DA1, 0, Wd4, Bd4, COARSE, 0, 1536, 1024, 64, 1, 0);
  k_point<<<(B2 * NP + 255) / 256, 256, 0, stream>>>(COARSE, PTF, PT16);

  // ---- refine buffers (reuse arenas) ----
  half_t* R_A  = ACT0;                                 // [16][512][<=512]
  half_t* R_B  = ACT0 + 16L * 512 * 512;
  half_t* CATX = ACT0 + 2 * 16L * 512 * 512;           // CAT2 / CAT3
  half_t* CAT4 = ACT0 + 3 * 16L * 512 * 512;           // [16][512][320]
  half_t* XL = ACT1;                                   // [16][512][256]
  half_t* X1 = ACT1 + 16L * 512 * 256;
  half_t* X2 = ACT1 + 2 * 16L * 512 * 256;
  half_t* X3 = ACT1 + 3 * 16L * 512 * 256;

  // r1: 3(->64) -> 256 -> 256 -> 256
  gemm(PT16, 64L * NP, Wr1a, Br1a, R_A, 256L * NP, 256, 64, NP, B2, 1);
  gemm(R_A, 256L * NP, Wr1b, Br1b, R_B, 256L * NP, 256, 256, NP, B2, 1);
  gemm(R_B, 256L * NP, Wr1c, Br1c, X1, 256L * NP, 256, 256, NP, B2, 0);
  k_chanmax<<<B2 * (256 / 64), 256, 0, stream>>>(X1, G16, NP, 256);
  {
    long total = (long)B2 * NP * 512;
    k_concat_bcast<<<(int)((total + 255) / 256), 256, 0, stream>>>(
        X1, G16, CATX, 256, NP, total);
  }
  // r2: 512 -> 256 -> 256 -> 256
  gemm(CATX, 512L * NP, Wr2a, Br2a, R_A, 256L * NP, 256, 512, NP, B2, 1);
  gemm(R_A, 256L * NP, Wr2b, Br2b, R_B, 256L * NP, 256, 256, NP, B2, 1);
  gemm(R_B, 256L * NP, Wr2c, Br2c, X2, 256L * NP, 256, 256, NP, B2, 0);
  // ball group + neighbor max
  k_ballquery<<<(B2 * NP + 255) / 256, 256, 0, stream>>>(PTF, IDX);
  k_gathermax<<<(int)(((long)B2 * NP * 256 + 255) / 256), 256, 0, stream>>>(
      X2, IDX, XL);
  {
    long total = (long)B2 * NP * 512;
    k_concat2<<<(int)((total + 255) / 256), 256, 0, stream>>>(
        X2, 256, 256, XL, 256, 256, CATX, 512, NP, total);
  }
  // r3: 512 -> 256 -> 256 -> 256
  gemm(CATX, 512L * NP, Wr3a, Br3a, R_A, 256L * NP, 256, 512, NP, B2, 1);
  gemm(R_A, 256L * NP, Wr3b, Br3b, R_B, 256L * NP, 256, 256, NP, B2, 1);
  gemm(R_B, 256L * NP, Wr3c, Br3c, X3, 256L * NP, 256, 256, NP, B2, 0);
  {
    long total = (long)B2 * NP * 320;
    k_concat2<<<(int)((total + 255) / 256), 256, 0, stream>>>(
        PT16, 64, 3, X3, 256, 256, CAT4, 320, NP, total);
  }
  // r4: 259(->320) -> 512 -> 512 -> 12(->64)
  gemm(CAT4, 320L * NP, Wr4a, Br4a, R_A, 512L * NP, 512, 320, NP, B2, 1);
  gemm(R_A, 512L * NP, Wr4b, Br4b, R_B, 512L * NP, 512, 512, NP, B2, 1);
  gemm(R_B, 512L * NP, Wr4c, Br4c, SHIFT, 64L * NP, 64, 512, NP, B2, 0);

  k_final<<<(B2 * NP * 12 + 255) / 256, 256, 0, stream>>>(
      PTF, SHIFT, (float*)d_out);
}
